// HydraAttention_v2_30562987278516
// MI455X (gfx1250) — matrix operand order repaired and checked by reading, not compiled, not run on hardware
//
#include <hip/hip_runtime.h>
#include <hip/hip_bf16.h>

// ---------------------------------------------------------------------------
// HydraAttention for MI455X (gfx1250).
// All contractions via v_wmma_f32_16x16x32_f16; fragments loaded with 16B
// vector LDS/global ops (B operands staged K-contiguous); conv weights staged
// to LDS via the Tensor Data Mover (tensor_load_to_lds + s_wait_tensorcnt).
// ---------------------------------------------------------------------------

typedef __attribute__((ext_vector_type(16))) _Float16 v16h;
typedef __attribute__((ext_vector_type(8)))  _Float16 v8h;
typedef __attribute__((ext_vector_type(4)))  _Float16 v4h;
typedef __attribute__((ext_vector_type(8)))  float    v8f;

#define BATCH 8
#define CH    64
#define HH    256
#define WW    256
#define NPIX  65536           // HH*WW

// ---- TDM (Tensor Data Mover) path, probe-verified builtin -----------------
#if __has_builtin(__builtin_amdgcn_tensor_load_to_lds)
#define HAVE_TDM 1
typedef __attribute__((ext_vector_type(4))) unsigned int v4u;
typedef __attribute__((ext_vector_type(8))) int          v8i;
typedef __attribute__((ext_vector_type(4))) int          v4i;

// 1-D TDM copy (nbytes multiple of 8). D# per CDNA5 ISA ch.8:
// group0: count=1 | lds_addr | global_addr[56:0] | type=2
// group1: data_size=3 (8B), tensor_dim0=tile_dim0=nbytes/8, tensor_dim1=1
__device__ __forceinline__ void tdm_load_1d(const void* g, void* lds,
                                            unsigned nbytes) {
  unsigned long long ga = (unsigned long long)g;
  unsigned lo = (unsigned)(unsigned long long)lds;   // LDS byte offset
  unsigned n8 = nbytes >> 3;                         // 8-byte elements
  v4u g0 = { 1u,                                     // count=1 (valid, user)
             lo,                                     // lds_addr
             (unsigned)ga,                           // global_addr[31:0]
             (unsigned)((ga >> 32) & 0x1FFFFFFu) | (2u << 30) };  // [56:32]|type=2
  v8i g1 = { (int)(3u << 16),                        // data_size=3 (8 bytes)
             (int)(n8 << 16),                        // tensor_dim0[15:0]
             (int)((n8 >> 16) | (1u << 16)),         // tensor_dim0[31:16]|dim1=1
             (int)(n8 << 16),                        // tile_dim0
             0,                                      // tile_dim1=0 (1D)
             (int)n8,                                // tensor_dim0_stride[31:0]
             0, 0 };
  v4i gz = { 0, 0, 0, 0 };
#if __clang_major__ >= 23
  v8i gz8 = { 0, 0, 0, 0, 0, 0, 0, 0 };
  __builtin_amdgcn_tensor_load_to_lds(g0, g1, gz, gz, gz8, 0);
#else
  __builtin_amdgcn_tensor_load_to_lds(g0, g1, gz, gz, 0);
#endif
}
__device__ __forceinline__ void tdm_wait0() {
#if __has_builtin(__builtin_amdgcn_s_wait_tensorcnt)
  __builtin_amdgcn_s_wait_tensorcnt(0);
#endif
}
#else
#define HAVE_TDM 0
#endif

__device__ __forceinline__ v8f wmma_ff(v16h a, v16h b, v8f c) {
  return __builtin_amdgcn_wmma_f32_16x16x32_f16(false, a, false, b, (short)0, c,
                                                false, false);
}

__device__ __forceinline__ v16h cat88(v8h lo, v8h hi) {
  return __builtin_shufflevector(lo, hi, 0, 1, 2, 3, 4, 5, 6, 7,
                                 8, 9, 10, 11, 12, 13, 14, 15);
}

// A: 16x32 (MxK) row-major, ld in halfs (multiple of 8, base 16B aligned).
// lane: M = lane&15 ; K runs [kb,kb+8) and [kb+16,kb+24), kb=(lane>>4)*8
__device__ __forceinline__ v16h loadA16(const _Float16* A, int ld) {
  int lane = threadIdx.x & 31;
  const _Float16* p = A + (lane & 15) * ld + ((lane >> 4) << 3);
  return cat88(*(const v8h*)p, *(const v8h*)(p + 16));
}

// B: 32x16 (KxN) staged K-contiguous: element (k,n) at Bt[n*ld + k].
// lane: N = lane&15 ; K runs [kb,kb+8) and [kb+16,kb+24), kb=(lane>>4)*8 -> 2x 16B loads
__device__ __forceinline__ v16h loadB16(const _Float16* Bt, int ld) {
  int lane = threadIdx.x & 31;
  const _Float16* p = Bt + (lane & 15) * ld + ((lane >> 4) << 3);
  return cat88(*(const v8h*)p, *(const v8h*)(p + 16));
}

// D/C 16x16 f32: VGPR r -> M = r + ((lane>>4)<<3), N = lane&15
__device__ __forceinline__ void storeD(float* D, int ld, v8f acc) {
  int lane = threadIdx.x & 31;
  int n  = lane & 15;
  int mo = (lane >> 4) << 3;
#pragma unroll
  for (int r = 0; r < 8; ++r) D[(mo + r) * ld + n] = acc[r];
}

__device__ __forceinline__ v8f biasAcc(const float* bias16) {
  int lane = threadIdx.x & 31;
  int mo = (lane >> 4) << 3;
  v8f c;
#pragma unroll
  for (int r = 0; r < 8; ++r) c[r] = bias16[mo + r];
  return c;
}

// ---------------------------------------------------------------------------
// kernel 0: zero f32 accumulators
// ---------------------------------------------------------------------------
__global__ void zero_kernel(float* p, int n) {
  int i = blockIdx.x * blockDim.x + threadIdx.x;
  if (i < n) p[i] = 0.0f;
}

// ---------------------------------------------------------------------------
// kernel 0b: convert conv weights [O][C][3][3] f32 -> f16 [sel][tap][o][c]
// ---------------------------------------------------------------------------
__global__ void wconv_kernel(const float* __restrict__ W1,
                             const float* __restrict__ W2,
                             _Float16* __restrict__ wh) {
  int i = blockIdx.x * blockDim.x + threadIdx.x;
  if (i >= 2 * 9 * 4096) return;
  int sel = i / (9 * 4096);
  int r   = i % (9 * 4096);
  int tap = r >> 12, o = (r >> 6) & 63, c = r & 63;
  const float* W = sel ? W2 : W1;
  wh[i] = (_Float16)W[(o * 64 + c) * 9 + tap];
}

// ---------------------------------------------------------------------------
// kernel 1: fused QKV GEMMs + L2-normalize + k_sum/value_sum partials
// grid = BATCH*(NPIX/256), block = 256 (8 waves)
// LDS: sxT[256][64]h | sw[3][64][64]h | sQ/sK/sV[64][256]f | srk,srq[256]f
// ---------------------------------------------------------------------------
__global__ void qkv_kernel(const float* __restrict__ x,
                           const float* __restrict__ Wq, const float* __restrict__ bq,
                           const float* __restrict__ Wk, const float* __restrict__ bk,
                           const float* __restrict__ Wv, const float* __restrict__ bv,
                           _Float16* __restrict__ Qn, _Float16* __restrict__ Kn,
                           _Float16* __restrict__ Vh,
                           float* __restrict__ k_sum, float* __restrict__ value_sum) {
  extern __shared__ char smem[];
  _Float16* sxT = (_Float16*)smem;                             // [256][64] 32768B
  _Float16* sw  = (_Float16*)(smem + 64 * 256 * 2);            // 24576B
  float*    sQ  = (float*)(smem + 64 * 256 * 2 + 3 * 64 * 64 * 2);
  float*    sK  = sQ + 64 * 256;
  float*    sV  = sK + 64 * 256;
  float*   srk  = sV + 64 * 256;                               // 256 f32
  float*   srq  = srk + 256;                                   // 256 f32

  const int blocksPerB = NPIX / 256;
  int b  = blockIdx.x / blocksPerB;
  int pb = (blockIdx.x % blocksPerB) * 256;
  int tid = threadIdx.x;

  // stage x tile transposed (px-major, K-contiguous for B fragments)
  for (int i = tid; i < 64 * 256; i += 256) {
    int c = i >> 8, p = i & 255;
    sxT[p * 64 + c] = (_Float16)x[((size_t)(b * CH + c)) * NPIX + pb + p];
  }
  for (int i = tid; i < 3 * 64 * 64; i += 256) {
    int sel = i >> 12, rc = i & 4095;
    const float* src = (sel == 0) ? Wq : (sel == 1) ? Wk : Wv;
    sw[i] = (_Float16)src[rc];
  }
  __syncthreads();

  // 3 mats x 4 m-tiles x 16 n-tiles = 192 WMMA tiles, K loop = 2
  int wave = tid >> 5;
  for (int t = wave; t < 192; t += 8) {
    int mat = t >> 6;
    int r = t & 63;
    int mt = r >> 4, nt = r & 15;
    const float* bp = (mat == 0) ? bq : (mat == 1) ? bk : bv;
    v8f acc = biasAcc(bp + mt * 16);
#pragma unroll
    for (int kk = 0; kk < 64; kk += 32) {
      v16h a  = loadA16(sw + mat * 4096 + (mt * 16) * 64 + kk, 64);
      v16h bb = loadB16(sxT + (nt * 16) * 64 + kk, 64);
      acc = wmma_ff(a, bb, acc);
    }
    float* dst = (mat == 0) ? sQ : (mat == 1) ? sK : sV;
    storeD(dst + (mt * 16) * 256 + nt * 16, 256, acc);
  }
  __syncthreads();

  // per-pixel L2 norms
  {
    int p = tid;
    float ssq = 0.f, ssk = 0.f;
#pragma unroll 4
    for (int c = 0; c < 64; ++c) {
      float q = sQ[c * 256 + p]; ssq += q * q;
      float k = sK[c * 256 + p]; ssk += k * k;
    }
    srq[p] = rsqrtf(fmaxf(ssq, 1e-24f));
    srk[p] = rsqrtf(fmaxf(ssk, 1e-24f));
  }
  __syncthreads();

  // vectorized (8B) global write-out of Qn/Kn/Vh
  for (int i = tid; i < 64 * 64; i += 256) {
    int c = i >> 6, p4 = (i & 63) * 4;
    v4h q, k, v;
#pragma unroll
    for (int j = 0; j < 4; ++j) {
      int p = p4 + j;
      q[j] = (_Float16)(sQ[c * 256 + p] * srq[p]);
      k[j] = (_Float16)(sK[c * 256 + p] * srk[p]);
      v[j] = (_Float16)(sV[c * 256 + p]);
    }
    size_t g = ((size_t)(b * CH + c)) * NPIX + pb + p4;
    *(v4h*)(Qn + g) = q;
    *(v4h*)(Kn + g) = k;
    *(v4h*)(Vh + g) = v;
  }

  // block-partial k_sum (normalized) and value_sum
  {
    int c = tid & 63;
    int seg = tid >> 6;
    float ka = 0.f, va = 0.f;
#pragma unroll 4
    for (int p = seg * 64; p < seg * 64 + 64; ++p) {
      ka += sK[c * 256 + p] * srk[p];
      va += sV[c * 256 + p];
    }
    atomicAdd(&k_sum[b * CH + c], ka);
    atomicAdd(&value_sum[b * CH + c], va);
  }
}

// ---------------------------------------------------------------------------
// kernel 2: agg_kv[b] = Kn_b (64xN) . Vh_b^T (Nx64), split-K over px chunks
// grid = BATCH*16 (4096-px chunks), block = 256 (8 waves)
// ---------------------------------------------------------------------------
__global__ void aggkv_kernel(const _Float16* __restrict__ Kn,
                             const _Float16* __restrict__ Vh,
                             float* __restrict__ agg_kv) {
  int b  = blockIdx.x >> 4;
  int pb = (blockIdx.x & 15) * 4096;
  int wave = threadIdx.x >> 5;

  __builtin_prefetch(Kn + ((size_t)(b * CH)) * NPIX + pb, 0, 1);
  __builtin_prefetch(Vh + ((size_t)(b * CH)) * NPIX + pb, 0, 1);

  for (int t = wave; t < 16; t += 8) {     // 4x4 tiles of the 64x64 result
    int mt = t >> 2, nt = t & 3;
    v8f acc = {};
    const _Float16* Ab = Kn + ((size_t)(b * CH + mt * 16)) * NPIX + pb;
    const _Float16* Bb = Vh + ((size_t)(b * CH + nt * 16)) * NPIX + pb;
#pragma unroll 4
    for (int kk = 0; kk < 4096; kk += 32) {
      v16h a  = loadA16(Ab + kk, NPIX);    // A[m][k] = Kn[ch][px]
      v16h bb = loadB16(Bb + kk, NPIX);    // B[k][n] = Vh[ch=n][px=k] (K-contig)
      acc = wmma_ff(a, bb, acc);
    }
    int lane = threadIdx.x & 31;
    int n = lane & 15, mo = (lane >> 4) << 3;
#pragma unroll
    for (int r = 0; r < 8; ++r)
      atomicAdd(&agg_kv[(b * CH + mt * 16 + mo + r) * 64 + nt * 16 + n], acc[r]);
  }
}

// ---------------------------------------------------------------------------
// kernel 3: agg_qkv = agg_kv^T.Qn -> denom/numerator -> attn = Wr.WV + br
// grid = BATCH*256 (256 px/block), block = 256 (8 waves)
// LDS: sQT[256][64]h | sAgT[64][64]h | sWr[64][64]h | sWV[64][256]f |
//      sWVhT[256][64]h | sDen[256]f
// ---------------------------------------------------------------------------
__global__ void attn_kernel(const _Float16* __restrict__ Qn,
                            const float* __restrict__ agg_kv,
                            const float* __restrict__ k_sum,
                            const float* __restrict__ value_sum,
                            const float* __restrict__ Wr, const float* __restrict__ br,
                            _Float16* __restrict__ attn) {
  extern __shared__ char smem[];
  _Float16* sQT   = (_Float16*)smem;             // [256][64] h
  _Float16* sAgT  = sQT + 64 * 256;              // [vc][kc] h
  _Float16* sWr   = sAgT + 64 * 64;              // [o][c] h
  float*    sWV   = (float*)(sWr + 64 * 64);     // [64][256] f
  _Float16* sWVhT = (_Float16*)(sWV + 64 * 256); // [256][64] h
  float*    sDen  = (float*)(sWVhT + 64 * 256);  // [256] f

  int b  = blockIdx.x >> 8;
  int pb = (blockIdx.x & 255) * 256;
  int tid = threadIdx.x;

  for (int i = tid; i < 64 * 256; i += 256) {
    int c = i >> 8, p = i & 255;
    sQT[p * 64 + c] = Qn[((size_t)(b * CH + c)) * NPIX + pb + p];
  }
  for (int i = tid; i < 4096; i += 256) {
    int vc = i >> 6, m = i & 63;                 // A[vc][m] = agg_kv[m][vc]
    sAgT[i] = (_Float16)agg_kv[(b * CH + m) * 64 + vc];
  }
  for (int i = tid; i < 4096; i += 256) sWr[i] = (_Float16)Wr[i];
  __syncthreads();

  // denom per pixel (K-contiguous LDS reads)
  {
    int p = tid;
    float acc = (float)NPIX;
#pragma unroll 4
    for (int m = 0; m < 64; ++m)
      acc += (float)sQT[p * 64 + m] * (k_sum[b * CH + m] + 1e-6f);
    sDen[p] = 1.0f / acc;
  }

  int wave = tid >> 5;
  // GEMM A: WV[vc][px] = sum_m aggT[vc][m] * Qn[m][px]
  for (int t = wave; t < 64; t += 8) {
    int mt = t >> 4, nt = t & 15;
    v8f acc = {};
#pragma unroll
    for (int kk = 0; kk < 64; kk += 32) {
      v16h a  = loadA16(sAgT + (mt * 16) * 64 + kk, 64);
      v16h bb = loadB16(sQT + (nt * 16) * 64 + kk, 64);
      acc = wmma_ff(a, bb, acc);
    }
    storeD(sWV + (mt * 16) * 256 + nt * 16, 256, acc);
  }
  __syncthreads();

  // numerator * denom -> f16, staged K-contiguous for the Wr GEMM
  for (int i = tid; i < 64 * 256; i += 256) {
    int c = i >> 8, p = i & 255;
    float wv = (value_sum[b * CH + c] + sWV[c * 256 + p]) * sDen[p];
    sWVhT[p * 64 + c] = (_Float16)wv;
  }
  __syncthreads();

  // GEMM B: attn = Wr . WV + br   (result back into sWV)
  for (int t = wave; t < 64; t += 8) {
    int mt = t >> 4, nt = t & 15;
    v8f acc = biasAcc(br + mt * 16);
#pragma unroll
    for (int kk = 0; kk < 64; kk += 32) {
      v16h a  = loadA16(sWr + (mt * 16) * 64 + kk, 64);
      v16h bb = loadB16(sWVhT + (nt * 16) * 64 + kk, 64);
      acc = wmma_ff(a, bb, acc);
    }
    storeD(sWV + (mt * 16) * 256 + nt * 16, 256, acc);
  }
  __syncthreads();

  // vectorized (8B) global write-out
  for (int i = tid; i < 64 * 64; i += 256) {
    int c = i >> 6, p4 = (i & 63) * 4;
    v4h o;
#pragma unroll
    for (int j = 0; j < 4; ++j) o[j] = (_Float16)sWV[c * 256 + p4 + j];
    *(v4h*)(attn + ((size_t)(b * CH + c)) * NPIX + pb + p4) = o;
  }
}

// ---------------------------------------------------------------------------
// kernel 4: 3x3 conv as implicit GEMM (K = 64*9), 16x16 output tile per block
// FUSE=true fuses  y*x + x  and writes f32 to d_out.
// LDS: sInT[324][64] f16 (channel-last 18x18 halo) | sW[9][64][64] f16
// weights arrive pre-converted [tap][o][c] f16 -> TDM DMA into LDS if avail
// ---------------------------------------------------------------------------
template <bool FUSE>
__global__ void conv3x3_kernel(const _Float16* __restrict__ in,
                               const _Float16* __restrict__ whT,
                               const float* __restrict__ bias,
                               _Float16* __restrict__ out_h,
                               float* __restrict__ out_f,
                               const float* __restrict__ xres) {
  extern __shared__ char smem[];
  _Float16* sInT = (_Float16*)smem;        // [324][64]
  _Float16* sW   = sInT + 324 * 64;        // [9][64][64] : [tap][o][c]

  int b    = blockIdx.x >> 8;
  int tile = blockIdx.x & 255;
  int ty = (tile >> 4) * 16, tx = (tile & 15) * 16;
  int tid = threadIdx.x;

  __builtin_prefetch(in + ((size_t)(b * CH)) * NPIX + ty * WW + tx, 0, 1);

#if HAVE_TDM
  if (tid < 32) tdm_load_1d(whT, sW, 9 * 4096 * 2);   // one DMA from wave 0
#else
  for (int i = tid; i < 9 * 4096; i += 256) sW[i] = whT[i];
#endif

  // halo 18x18, channel-last (K-contiguous for B fragments), zero padded
  for (int i = tid; i < 64 * 324; i += 256) {
    int c = i / 324, s = i % 324;
    int iy = ty + (s / 18) - 1;
    int ix = tx + (s % 18) - 1;
    _Float16 v = (_Float16)0.0f;
    if (iy >= 0 && iy < HH && ix >= 0 && ix < WW)
      v = in[((size_t)(b * CH + c)) * NPIX + iy * WW + ix];
    sInT[s * 64 + c] = v;
  }
#if HAVE_TDM
  if (tid < 32) tdm_wait0();               // wave 0 drains TENSORcnt, then barrier
#endif
  __syncthreads();

  int wave = tid >> 5;
  // n-tile nt is exactly output row nt: p = nt*16 + n -> (py,px) = (nt, n)
  for (int t = wave; t < 64; t += 8) {     // 4 m-tiles x 16 row-tiles
    int mt = t >> 4, nt = t & 15;
    v8f acc = biasAcc(bias + mt * 16);
#pragma unroll
    for (int tap = 0; tap < 9; ++tap) {
      int ky = tap / 3, kx = tap % 3;
      const _Float16* bbase = sInT + ((nt + ky) * 18 + kx) * 64;
#pragma unroll
      for (int kk = 0; kk < 64; kk += 32) {
        v16h a  = loadA16(sW + tap * 4096 + (mt * 16) * 64 + kk, 64);
        v16h bb = loadB16(bbase + kk, 64);
        acc = wmma_ff(a, bb, acc);
      }
    }
    int lane = tid & 31;
    int n = lane & 15, mo = (lane >> 4) << 3;
#pragma unroll
    for (int r = 0; r < 8; ++r) {
      int o = mt * 16 + mo + r;
      size_t g = ((size_t)(b * CH + o)) * NPIX + (size_t)(ty + nt) * WW + (tx + n);
      if (FUSE) {
        float xr = xres[g];
        out_f[g] = acc[r] * xr + xr;       // conv2(.)*x + x
      } else {
        out_h[g] = (_Float16)acc[r];
      }
    }
  }
}

// ---------------------------------------------------------------------------
extern "C" void kernel_launch(void* const* d_in, const int* in_sizes, int n_in,
                              void* d_out, int out_size, void* d_ws, size_t ws_size,
                              hipStream_t stream) {
  const float* x  = (const float*)d_in[0];
  const float* Wq = (const float*)d_in[1];
  const float* bq = (const float*)d_in[2];
  const float* Wk = (const float*)d_in[3];
  const float* bk = (const float*)d_in[4];
  const float* Wv = (const float*)d_in[5];
  const float* bv = (const float*)d_in[6];
  const float* Wr = (const float*)d_in[7];
  const float* br = (const float*)d_in[8];
  const float* W1 = (const float*)d_in[9];
  const float* b1 = (const float*)d_in[10];
  const float* W2 = (const float*)d_in[11];
  const float* b2 = (const float*)d_in[12];
  float* out = (float*)d_out;

  char* ws = (char*)d_ws;
  const size_t S = (size_t)BATCH * CH * NPIX * sizeof(_Float16);  // 64 MB
  _Float16* Qn = (_Float16*)(ws);
  _Float16* Kn = (_Float16*)(ws + S);
  _Float16* Vh = (_Float16*)(ws + 2 * S);
  _Float16* c1 = (_Float16*)(ws + 3 * S);      // attn (f16)
  _Float16* t1 = (_Float16*)(ws + 4 * S);      // after first 3x3 conv
  float* k_sum     = (float*)(ws + 5 * S);     // [B,64]
  float* value_sum = k_sum + BATCH * CH;       // [B,64]
  float* agg_kv    = value_sum + BATCH * CH;   // [B,64,64]
  _Float16* wh     = (_Float16*)(agg_kv + BATCH * CH * CH);  // [2][9][64][64]

  const int nAccum = BATCH * CH * 2 + BATCH * CH * CH;
  zero_kernel<<<(nAccum + 255) / 256, 256, 0, stream>>>(k_sum, nAccum);
  wconv_kernel<<<(2 * 9 * 4096 + 255) / 256, 256, 0, stream>>>(W1, W2, wh);

  size_t smem1 = (size_t)64 * 256 * 2 + 3 * 64 * 64 * 2 + 3 * 64 * 256 * 4 +
                 2 * 256 * 4;
  qkv_kernel<<<BATCH * (NPIX / 256), 256, smem1, stream>>>(
      x, Wq, bq, Wk, bk, Wv, bv, Qn, Kn, Vh, k_sum, value_sum);

  aggkv_kernel<<<BATCH * 16, 256, 0, stream>>>(Kn, Vh, agg_kv);

  size_t smem3 = (size_t)64 * 256 * 2 + 64 * 64 * 2 + 64 * 64 * 2 +
                 64 * 256 * 4 + 64 * 256 * 2 + 256 * 4;
  attn_kernel<<<BATCH * 256, 256, smem3, stream>>>(
      Qn, agg_kv, k_sum, value_sum, Wr, br, c1);

  size_t smem4 = (size_t)324 * 64 * 2 + 9 * 64 * 64 * 2;
  conv3x3_kernel<false><<<BATCH * 256, 256, smem4, stream>>>(
      c1, wh, b1, t1, nullptr, nullptr);
  conv3x3_kernel<true><<<BATCH * 256, 256, smem4, stream>>>(
      t1, wh + 9 * 4096, b2, nullptr, out, x);
}
